// Block_6897717477392
// MI455X (gfx1250) — matrix operand order repaired and checked by reading, not compiled, not run on hardware
//
#include <hip/hip_runtime.h>
#include <hip/hip_bf16.h>
#include <stdint.h>

#define EMBED 2048
#define HEADS 16
#define HEAD_DIM 128
#define FF_HID 8192
#define BATCH 2
#define SEQ 2048
#define NTOK (BATCH*SEQ)   /* 4096 tokens */

typedef __attribute__((ext_vector_type(16))) __bf16 v16bf;
typedef __attribute__((ext_vector_type(8)))  float  v8f;
typedef __attribute__((ext_vector_type(8)))  unsigned short us8;
typedef __attribute__((ext_vector_type(4)))  float  f4;

union FragU { v16bf v; us8 u[2]; };

__device__ __forceinline__ unsigned short f2bf(float f) {
    unsigned u = __float_as_uint(f);
    u += 0x7FFFu + ((u >> 16) & 1u);        // round-to-nearest-even
    return (unsigned short)(u >> 16);
}
__device__ __forceinline__ float bf2f(unsigned short h) {
    return __uint_as_float(((unsigned)h) << 16);
}

// async memory->LDS copy of 16 bytes per lane (CDNA5, tracked by ASYNCcnt)
__device__ __forceinline__ void async_copy_b128(uint32_t lds_byte_addr,
                                                uint32_t gl_byte_off,
                                                uint64_t gl_base) {
    asm volatile("global_load_async_to_lds_b128 %0, %1, %2"
                 :: "v"(lds_byte_addr), "v"(gl_byte_off), "s"(gl_base)
                 : "memory");
}
__device__ __forceinline__ void async_wait0() {
    asm volatile("s_wait_asynccnt 0" ::: "memory");
}

// ---------------------------------------------------------------------------
// RMSNorm: fp32 in -> bf16 out (one block per token)
// ---------------------------------------------------------------------------
__global__ __launch_bounds__(256) void rmsnorm_bf16_kernel(
        const float* __restrict__ x, const float* __restrict__ scale,
        unsigned short* __restrict__ out) {
    __shared__ float red[256];
    const int t = blockIdx.x, tid = threadIdx.x;
    const float* xr = x + (size_t)t * EMBED;
    float v[8]; float ss = 0.f;
    #pragma unroll
    for (int i = 0; i < 8; ++i) { v[i] = xr[i*256 + tid]; ss += v[i]*v[i]; }
    red[tid] = ss; __syncthreads();
    for (int o = 128; o > 0; o >>= 1) {
        if (tid < o) red[tid] += red[tid + o];
        __syncthreads();
    }
    const float r = rsqrtf(red[0] * (1.0f/EMBED) + 1e-6f);
    unsigned short* orow = out + (size_t)t * EMBED;
    #pragma unroll
    for (int i = 0; i < 8; ++i) {
        const int c = i*256 + tid;
        orow[c] = f2bf(v[i] * r * scale[c]);
    }
}

// ---------------------------------------------------------------------------
// GEMM: C[M,N] = A[M,K](bf16, row-major) * B[K,N](fp32, converted on the fly)
// EPI 0: C stored bf16.  EPI 1: C stored fp32 with residual add.
// Block: 256 thr = 8 waves, tile 128(M)x128(N); per wave 32x64 = 2x4 WMMA
// tiles; K-step 32. A tile staged with global_load_async_to_lds_b128.
// ---------------------------------------------------------------------------
template<int EPI>
__global__ __launch_bounds__(256) void gemm_wmma_kernel(
        const unsigned short* __restrict__ A, const float* __restrict__ B,
        void* __restrict__ Cout, const float* __restrict__ Res,
        int M, int N, int K) {
    __shared__ __align__(16) unsigned short As[128*40];  // [m][k], pad 40
    __shared__ __align__(16) unsigned short Bs[128*40];  // [n][k] transposed
    const int tid  = threadIdx.x;
    const int lane = tid & 31, wave = tid >> 5;
    const int wm = wave & 3, wn = wave >> 2;        // 4 x 2 wave grid
    const int bm = blockIdx.y * 128, bn = blockIdx.x * 128;
    const int hl  = (lane < 16) ? 0 : 1;
    const int l16 = lane & 15;

    const uint32_t asBase = (uint32_t)(uintptr_t)(&As[0]);
    const uint64_t aBase  = (uint64_t)(uintptr_t)A;

    v8f zero = {0.f,0.f,0.f,0.f,0.f,0.f,0.f,0.f};
    v8f acc[2][4];
    #pragma unroll
    for (int i=0;i<2;++i)
        #pragma unroll
        for (int j=0;j<4;++j) acc[i][j] = zero;

    for (int kt = 0; kt < K; kt += 32) {
        __syncthreads();
        // ---- stage A (128x32 bf16) via async global->LDS b128 copies
        #pragma unroll
        for (int p = 0; p < 2; ++p) {
            const int idx = p*256 + tid;
            const int row = idx >> 2, cg = (idx & 3) * 8;
            const uint32_t lds = asBase + (uint32_t)(row*40 + cg)*2u;
            const uint32_t off = (uint32_t)(((size_t)(bm+row)*K + kt + cg)*2u);
            async_copy_b128(lds, off, aBase);
        }
        // ---- stage B (32x128 fp32 -> bf16, transposed to [n][k])
        {
            const int k = tid >> 3, n0 = (tid & 7) * 16;
            const float* src = &B[(size_t)(kt+k)*N + bn + n0];
            #pragma unroll
            for (int q4 = 0; q4 < 4; ++q4) {
                f4 b0 = *(const f4*)(src + q4*4);
                #pragma unroll
                for (int i=0;i<4;++i) Bs[(n0+q4*4+i)*40 + k] = f2bf(b0[i]);
            }
        }
        async_wait0();
        __syncthreads();

        // ---- fragments (ISA A/B VGPR layouts) + 2x4 WMMA
        FragU a[2], b[4];
        #pragma unroll
        for (int i=0;i<2;++i) {
            const int row = wm*32 + i*16 + l16;
            const int kb = hl * 8;   // lanes16-31 start at K=8
            a[i].u[0] = *(const us8*)&As[row*40 + kb];        // K kb..kb+7
            a[i].u[1] = *(const us8*)&As[row*40 + kb + 16];   // K kb+16..kb+23
        }
        #pragma unroll
        for (int j=0;j<4;++j) {
            const int n = wn*64 + j*16 + l16;
            const int kb = hl * 8;   // lanes16-31 start at K=8
            b[j].u[0] = *(const us8*)&Bs[n*40 + kb];          // K kb..kb+7
            b[j].u[1] = *(const us8*)&Bs[n*40 + kb + 16];     // K kb+16..kb+23
        }
        #pragma unroll
        for (int i=0;i<2;++i)
            #pragma unroll
            for (int j=0;j<4;++j)
                acc[i][j] = __builtin_amdgcn_wmma_f32_16x16x32_bf16(
                    false, a[i].v, false, b[j].v, (short)0, acc[i][j], false, false);
    }

    // ---- epilogue (C layout: VGPR r -> row r / r+8, lane%16 -> col)
    #pragma unroll
    for (int i=0;i<2;++i)
      #pragma unroll
      for (int j=0;j<4;++j)
        #pragma unroll
        for (int r=0;r<8;++r) {
            const int row = bm + wm*32 + i*16 + r + hl*8;
            const int col = bn + wn*64 + j*16 + l16;
            const size_t o = (size_t)row*N + col;
            const float vv = acc[i][j][r];
            if (EPI == 0) ((unsigned short*)Cout)[o] = f2bf(vv);
            else          ((float*)Cout)[o] = vv + Res[o];
        }
}

// ---------------------------------------------------------------------------
// Flash attention, causal. qkv: [token][h*384 + {q:0-127,k:128-255,v:256-383}]
// bf16. out: [token][h*128+d] bf16. Block = 8 waves x 16 q-rows = 128 q rows,
// grid (S/128, B*H). Key tiles of 32, online softmax, all GEMMs via WMMA.
// K tile staged with async global->LDS copies; V staged transposed manually.
// ---------------------------------------------------------------------------
__global__ __launch_bounds__(256) void flash_attn_kernel(
        const unsigned short* __restrict__ qkv,
        unsigned short* __restrict__ out) {
    __shared__ __align__(16) unsigned short Ks[32*136];   // [key][d], pad 136
    __shared__ __align__(16) unsigned short Vs[128*40];   // [d][key], pad 40
    __shared__ __align__(16) unsigned short Ps[8*16*40];  // per-wave P scratch

    const int tid  = threadIdx.x;
    const int lane = tid & 31, wave = tid >> 5;
    const int bh = blockIdx.y, b = bh >> 4, h = bh & 15;
    const int qb = blockIdx.x * 128;
    const int qbase = qb + wave * 16;
    const int hl  = (lane < 16) ? 0 : 1;
    const int l16 = lane & 15;
    const size_t tokStride = 3*HEAD_DIM*HEADS;     // 6144
    const size_t headOff   = (size_t)h * 3*HEAD_DIM;

    const uint32_t ksBase  = (uint32_t)(uintptr_t)(&Ks[0]);
    const uint64_t qkvBase = (uint64_t)(uintptr_t)qkv;

    // Q fragments: 16 rows x 128 d as 4 A-frags (d-chunks of 32)
    FragU qf[4];
    {
        const int q = qbase + l16;
        const unsigned short* qrow = qkv + ((size_t)(b*SEQ + q))*tokStride + headOff;
        const int kb = hl*8;
        #pragma unroll
        for (int dc = 0; dc < 4; ++dc) {
            qf[dc].u[0] = *(const us8*)&qrow[dc*32 + kb];
            qf[dc].u[1] = *(const us8*)&qrow[dc*32 + kb + 16];
        }
    }

    v8f zero = {0.f,0.f,0.f,0.f,0.f,0.f,0.f,0.f};
    float m[8], l[8];
    v8f O[8];
    #pragma unroll
    for (int r=0;r<8;++r) { m[r] = -3.0e38f; l[r] = 0.f; }
    #pragma unroll
    for (int n=0;n<8;++n) O[n] = zero;

    const int numj = qb/32 + 4;                    // uniform across block
    const float SCL = 0.08838834764831845f;        // 1/sqrt(128)

    for (int j = 0; j < numj; ++j) {
        __syncthreads();
        // stage K (async b128 copies) and V transposed (shared by all waves)
        #pragma unroll
        for (int p = 0; p < 2; ++p) {
            const int idx = p*256 + tid;
            const int key = idx >> 4, seg = (idx & 15) * 8;
            const size_t rowOff = ((size_t)(b*SEQ + j*32 + key))*tokStride + headOff;
            const uint32_t lds = ksBase + (uint32_t)(key*136 + seg)*2u;
            async_copy_b128(lds, (uint32_t)((rowOff + 128 + seg)*2u), qkvBase);
            us8 vv = *(const us8*)&qkv[rowOff + 256 + seg];
            #pragma unroll
            for (int i=0;i<8;++i) Vs[(seg+i)*40 + key] = vv[i];
        }
        async_wait0();
        __syncthreads();

        // logits: S(16q x 32key) = Q(16x128) * K^T, 2 key subtiles x 4 d-chunks
        v8f sT[2] = { zero, zero };
        #pragma unroll
        for (int sub = 0; sub < 2; ++sub)
            #pragma unroll
            for (int dc = 0; dc < 4; ++dc) {
                FragU kf;
                const int key = sub*16 + l16;        // B-frag column = key
                const int kb2 = hl * 8;              // d within chunk
                kf.u[0] = *(const us8*)&Ks[key*136 + dc*32 + kb2];
                kf.u[1] = *(const us8*)&Ks[key*136 + dc*32 + kb2 + 16];
                sT[sub] = __builtin_amdgcn_wmma_f32_16x16x32_bf16(
                    false, qf[dc].v, false, kf.v, (short)0, sT[sub], false, false);
            }

        // online softmax (row reductions across the 16-lane half)
        unsigned short* Pw = &Ps[wave*640];
        #pragma unroll
        for (int r = 0; r < 8; ++r) {
            const int q  = qbase + r + hl*8;
            const int k0 = j*32 + l16;
            float s0 = (k0      <= q) ? sT[0][r]*SCL : -1.0e30f;
            float s1 = (k0 + 16 <= q) ? sT[1][r]*SCL : -1.0e30f;
            float mx = fmaxf(s0, s1);
            #pragma unroll
            for (int o = 8; o >= 1; o >>= 1) mx = fmaxf(mx, __shfl_xor(mx, o, 32));
            const float mnew  = fmaxf(m[r], mx);
            const float alpha = __expf(m[r] - mnew);
            const float p0 = __expf(s0 - mnew);
            const float p1 = __expf(s1 - mnew);
            float rs = p0 + p1;
            #pragma unroll
            for (int o = 8; o >= 1; o >>= 1) rs += __shfl_xor(rs, o, 32);
            l[r] = l[r]*alpha + rs;
            m[r] = mnew;
            #pragma unroll
            for (int n=0;n<8;++n) O[n][r] *= alpha;
            const int row = r + hl*8;
            Pw[row*40 + l16]      = f2bf(p0);
            Pw[row*40 + l16 + 16] = f2bf(p1);
        }
        // per-wave LDS RAW fence before reading P in A-layout
        asm volatile("s_wait_dscnt 0" ::: "memory");

        FragU pf;
        {
            const int kb = hl*8;
            pf.u[0] = *(const us8*)&Pw[l16*40 + kb];
            pf.u[1] = *(const us8*)&Pw[l16*40 + kb + 16];
        }
        // O(16x128) += P(16x32) * V(32x128)
        #pragma unroll
        for (int n = 0; n < 8; ++n) {
            FragU vf;
            const int d = n*16 + l16;
            const int kb2 = hl*8;
            vf.u[0] = *(const us8*)&Vs[d*40 + kb2];
            vf.u[1] = *(const us8*)&Vs[d*40 + kb2 + 16];
            O[n] = __builtin_amdgcn_wmma_f32_16x16x32_bf16(
                false, pf.v, false, vf.v, (short)0, O[n], false, false);
        }
    }

    // finalize and store bf16 [token][h*128+d]
    #pragma unroll
    for (int r = 0; r < 8; ++r) {
        const float inv = 1.0f / l[r];
        const int q = qbase + r + hl*8;
        unsigned short* orow = out + ((size_t)(b*SEQ + q))*EMBED + h*HEAD_DIM;
        #pragma unroll
        for (int n = 0; n < 8; ++n)
            orow[n*16 + l16] = f2bf(O[n][r] * inv);
    }
}

// ---------------------------------------------------------------------------
// SwiGLU combine: gate <- gelu_tanh(gate) * up   (bf16 in-place over gate)
// ---------------------------------------------------------------------------
__global__ __launch_bounds__(256) void swiglu_kernel(
        unsigned short* __restrict__ gate, const unsigned short* __restrict__ up) {
    const size_t i = (size_t)blockIdx.x * 256 + threadIdx.x;
    const float g = bf2f(gate[i]);
    const float u = bf2f(up[i]);
    const float t = tanhf(0.7978845608028654f * (g + 0.044715f * g*g*g));
    gate[i] = f2bf(0.5f * g * (1.f + t) * u);
}

// ---------------------------------------------------------------------------
extern "C" void kernel_launch(void* const* d_in, const int* in_sizes, int n_in,
                              void* d_out, int out_size, void* d_ws, size_t ws_size,
                              hipStream_t stream) {
    (void)in_sizes; (void)n_in; (void)out_size; (void)ws_size;
    const float* x      = (const float*)d_in[0];
    const float* w_qkv  = (const float*)d_in[1];   // [2048][6144]
    const float* w_out  = (const float*)d_in[2];   // [2048][2048] (h*128+d major)
    const float* w_gate = (const float*)d_in[3];   // [2048][8192]
    const float* w_up   = (const float*)d_in[4];   // [2048][8192]
    const float* w_down = (const float*)d_in[5];   // [8192][2048]
    const float* scale1 = (const float*)d_in[6];
    const float* scale2 = (const float*)d_in[7];

    char* ws = (char*)d_ws;
    unsigned short* xn   = (unsigned short*)ws;  ws += (size_t)NTOK*EMBED*2;   // 16.8MB
    unsigned short* qkvb = (unsigned short*)ws;  ws += (size_t)NTOK*6144*2;    // 50.3MB
    unsigned short* attn = (unsigned short*)ws;  ws += (size_t)NTOK*EMBED*2;   // 16.8MB
    float*          h1   = (float*)ws;           ws += (size_t)NTOK*EMBED*4;   // 33.6MB
    unsigned short* gate = (unsigned short*)ws;  ws += (size_t)NTOK*FF_HID*2;  // 67.1MB
    unsigned short* up   = (unsigned short*)ws;  ws += (size_t)NTOK*FF_HID*2;  // 67.1MB

    const dim3 blk(256);
    // 1) xn = rmsnorm(x, scale1)  (bf16)
    rmsnorm_bf16_kernel<<<NTOK, blk, 0, stream>>>(x, scale1, xn);
    // 2) qkv = xn @ w_qkv  (bf16)
    gemm_wmma_kernel<0><<<dim3(6144/128, NTOK/128), blk, 0, stream>>>(
        xn, w_qkv, qkvb, nullptr, NTOK, 6144, EMBED);
    // 3) attn = flash_attention(qkv)  (bf16)
    flash_attn_kernel<<<dim3(SEQ/128, BATCH*HEADS), blk, 0, stream>>>(qkvb, attn);
    // 4) h1 = attn @ w_out + x  (fp32)
    gemm_wmma_kernel<1><<<dim3(EMBED/128, NTOK/128), blk, 0, stream>>>(
        attn, w_out, h1, x, NTOK, EMBED, EMBED);
    // 5) xn = rmsnorm(h1, scale2)
    rmsnorm_bf16_kernel<<<NTOK, blk, 0, stream>>>(h1, scale2, xn);
    // 6) gate = xn @ w_gate ; up = xn @ w_up  (bf16)
    gemm_wmma_kernel<0><<<dim3(FF_HID/128, NTOK/128), blk, 0, stream>>>(
        xn, w_gate, gate, nullptr, NTOK, FF_HID, EMBED);
    gemm_wmma_kernel<0><<<dim3(FF_HID/128, NTOK/128), blk, 0, stream>>>(
        xn, w_up, up, nullptr, NTOK, FF_HID, EMBED);
    // 7) gate = gelu(gate) * up
    swiglu_kernel<<<((size_t)NTOK*FF_HID)/256, blk, 0, stream>>>(gate, up);
    // 8) out = gate @ w_down + h1  (fp32)
    gemm_wmma_kernel<1><<<dim3(EMBED/128, NTOK/128), blk, 0, stream>>>(
        gate, w_down, (float*)d_out, h1, NTOK, EMBED, FF_HID);
}
